// Model_70540542869553
// MI455X (gfx1250) — matrix operand order repaired and checked by reading, not compiled, not run on hardware
//
#include <hip/hip_runtime.h>
#include <stdint.h>

typedef _Float16 half8 __attribute__((ext_vector_type(8)));
typedef _Float16 v16h  __attribute__((ext_vector_type(16)));
typedef float    v8f   __attribute__((ext_vector_type(8)));
typedef float    f4    __attribute__((ext_vector_type(4)));

#define CIN   64
#define COUT  64
#define KTAPS 9
#define KDIM  576      // CIN * KTAPS
#define ASTR  584      // padded LDS row stride in halfs (292 dwords % 64 banks = 36 -> conflict-free b128)
#define HOUT  128
#define WOUT  128
#define HIN   130
#define WIN   130
#define MTILE 32       // output pixels per block
#define THREADS 256    // 8 wave32
#define NPAIR (MTILE * KTAPS)   // 288 (pixel,tap) records

// --- pre-kernel: weight f32 [Cout][Cin][3][3] -> f16 [Cout][576] packed (flat order)
__global__ void dcn_w16_cvt(const float* __restrict__ w, _Float16* __restrict__ w16, int n) {
    int i = blockIdx.x * blockDim.x + threadIdx.x;
    if (i < n) w16[i] = (_Float16)w[i];
}

__launch_bounds__(THREADS)
__global__ void dcn_main(const float* __restrict__ x,
                         const float* __restrict__ offs,
                         const float* __restrict__ msk,
                         const _Float16* __restrict__ w16,
                         float* __restrict__ out)
{
    __shared__ _Float16 sA[MTILE * ASTR];   // 37,376 B : im2col tile (f16, padded rows)
    __shared__ _Float16 sB[COUT * ASTR];    // 74,752 B : full weight matrix (f16, padded rows)
    __shared__ unsigned sPair[NPAIR * 8];   //  9,216 B : per-(pixel,tap) {w00..w11, i00..i11}

    const int tile = blockIdx.x;            // 0..2047
    const int p0   = tile * MTILE;
    const int b    = p0 >> 14;
    const int hw0  = p0 & 16383;
    const int h    = hw0 >> 7;
    const int w0   = hw0 & 127;             // 32-aligned, row-contiguous tile

    const int t = threadIdx.x;

    // ---- stage 0: async-stage the whole f16 weight into LDS (overlaps all sampling work) ----
    // Row-repack: global stride 1152 B (packed) -> LDS stride 1168 B (padded).
    {
        const int r = t >> 2;               // weight row 0..63
        const int q = t & 3;                // quarter-row: 288 B each (18 x 16 B)
        unsigned ldsb = (unsigned)(size_t)(&sB[0]) + (unsigned)(r * (ASTR * 2) + q * 288);
        unsigned goff = (unsigned)(r * (KDIM * 2) + q * 288);
        uint64_t wp = (uint64_t)(const void*)w16;
        #pragma unroll
        for (int i = 0; i < 18; ++i) {
            asm volatile("global_load_async_to_lds_b128 %0, %1, %2 offset:%3"
                         :: "v"(ldsb + i * 16), "v"(goff + i * 16), "s"(wp), "i"(0)
                         : "memory");
        }
    }

    // ---- stage 1: per-(pixel,tap) bilinear records, computed once (not once per channel group) ----
    for (int pr = t; pr < NPAIR; pr += THREADS) {
        const int j  = pr >> 5;             // tap 0..8
        const int m  = pr & 31;             // pixel in tile
        const int ki = j / 3, kj = j % 3;
        const int wp = w0 + m;
        float dy = offs[((b * 2 * KTAPS + 2 * j    ) * HOUT + h) * WOUT + wp];
        float dx = offs[((b * 2 * KTAPS + 2 * j + 1) * HOUT + h) * WOUT + wp];
        float mk = msk[((b * KTAPS + j) * HOUT + h) * WOUT + wp];
        float py = dy + (float)ki + (float)h;
        float px = dx + (float)kj + (float)wp;
        float y0f = floorf(py), x0f = floorf(px);
        float ly = py - y0f, lx = px - x0f;
        int y0 = (int)y0f, x0 = (int)x0f;
        int y1 = y0 + 1,   x1 = x0 + 1;
        float vy0 = (y0 >= 0 && y0 < HIN) ? 1.f : 0.f;
        float vy1 = (y1 >= 0 && y1 < HIN) ? 1.f : 0.f;
        float vx0 = (x0 >= 0 && x0 < WIN) ? 1.f : 0.f;
        float vx1 = (x1 >= 0 && x1 < WIN) ? 1.f : 0.f;
        float w00 = (1.f - ly) * (1.f - lx) * mk * vy0 * vx0;
        float w01 = (1.f - ly) * lx         * mk * vy0 * vx1;
        float w10 = ly         * (1.f - lx) * mk * vy1 * vx0;
        float w11 = ly         * lx         * mk * vy1 * vx1;
        int cy0 = min(max(y0, 0), HIN - 1), cy1 = min(max(y1, 0), HIN - 1);
        int cx0 = min(max(x0, 0), WIN - 1), cx1 = min(max(x1, 0), WIN - 1);
        unsigned* rec = &sPair[pr * 8];
        rec[0] = __float_as_uint(w00);
        rec[1] = __float_as_uint(w01);
        rec[2] = __float_as_uint(w10);
        rec[3] = __float_as_uint(w11);
        rec[4] = (unsigned)(cy0 * WIN + cx0);
        rec[5] = (unsigned)(cy0 * WIN + cx1);
        rec[6] = (unsigned)(cy1 * WIN + cx0);
        rec[7] = (unsigned)(cy1 * WIN + cx1);
    }
    __syncthreads();

    // ---- stage 2: gather + modulate -> sA[m][c*9+j] (each thread: 1 pixel x 8 channels x 9 taps) ----
    {
        const int m = t & 31;
        const int s = t >> 5;
        #pragma unroll
        for (int j = 0; j < KTAPS; ++j) {
            const unsigned* rec = &sPair[(j * 32 + m) * 8];
            float w00 = __uint_as_float(rec[0]);
            float w01 = __uint_as_float(rec[1]);
            float w10 = __uint_as_float(rec[2]);
            float w11 = __uint_as_float(rec[3]);
            int i00 = (int)rec[4], i01 = (int)rec[5], i10 = (int)rec[6], i11 = (int)rec[7];
            #pragma unroll
            for (int cc = 0; cc < 8; ++cc) {
                int c = s * 8 + cc;
                const float* xp = x + (size_t)(b * CIN + c) * (HIN * WIN);
                float val = w00 * xp[i00] + w01 * xp[i01] + w10 * xp[i10] + w11 * xp[i11];
                sA[m * ASTR + c * KTAPS + j] = (_Float16)val;
            }
        }
    }
    asm volatile("s_wait_asynccnt 0x0" ::: "memory");   // weight fully resident in LDS
    __syncthreads();

    // ---- stage 3: GEMM on matrix cores: 8 waves = 2(M) x 4(N) tiles of 16x16, K=576 ----
    const int lane = t & 31;
    const int wv   = t >> 5;
    const int mt   = wv >> 2;
    const int nt   = wv & 3;
    const int hseg = lane >> 4;
    const int l15  = lane & 15;
    const int mA   = mt * 16 + l15;
    const int n    = nt * 16 + l15;

    v8f acc = {};
    const _Float16* sArow = sA + mA * ASTR + hseg * 16;   // A layout: lane->M, contiguous 16-K per half
    const _Float16* sBrow = sB + n  * ASTR + hseg * 16;   // B layout: lane->N, contiguous 16-K per half

    #pragma unroll
    for (int kc = 0; kc < 18; ++kc) {
        half8 alo = *(const half8*)(sArow + kc * 32);
        half8 ahi = *(const half8*)(sArow + kc * 32 + 8);
        v16h a = __builtin_shufflevector(alo, ahi, 0,1,2,3,4,5,6,7,8,9,10,11,12,13,14,15);
        v16h bf = *(const v16h*)(sBrow + kc * 32);
        acc = __builtin_amdgcn_wmma_f32_16x16x32_f16(
                  false, a, false, bf, (short)0, acc, false, false);
    }

    // ---- store D: 8 consecutive-w floats per lane -> two coalesced float4 stores ----
    size_t obase = (size_t)(b * COUT + n) * (HOUT * WOUT)
                 + (size_t)h * WOUT + w0 + mt * 16 + hseg * 8;
    f4 lo4 = {acc[0], acc[1], acc[2], acc[3]};
    f4 hi4 = {acc[4], acc[5], acc[6], acc[7]};
    *(f4*)(out + obase)     = lo4;
    *(f4*)(out + obase + 4) = hi4;
}

extern "C" void kernel_launch(void* const* d_in, const int* in_sizes, int n_in,
                              void* d_out, int out_size, void* d_ws, size_t ws_size,
                              hipStream_t stream) {
    const float* x      = (const float*)d_in[0];
    const float* offs   = (const float*)d_in[1];
    const float* msk    = (const float*)d_in[2];
    const float* weight = (const float*)d_in[3];
    float* out = (float*)d_out;
    _Float16* w16 = (_Float16*)d_ws;        // 64*576 f16 = 73,728 B scratch

    const int nw = COUT * KDIM;
    dcn_w16_cvt<<<(nw + 255) / 256, 256, 0, stream>>>(weight, w16, nw);
    dcn_main<<<2048, THREADS, 0, stream>>>(x, offs, msk, w16, out);
    (void)in_sizes; (void)n_in; (void)out_size; (void)ws_size;
}
